// Seq2SeqLSTM_70635032150030
// MI455X (gfx1250) — matrix operand order repaired and checked by reading, not compiled, not run on hardware
//
#include <hip/hip_runtime.h>

#define N_   4096
#define T_   64
#define FIN_ 8
#define E_   256
#define H_   512
#define FW_  32
#define G_   2048          // 4*H

#define BM   16            // batch rows per block
#define NTHREADS 512       // 16 waves (wave32)

#define K0   (E_ + H_)     // 768  : [emb | h0]
#define K1   (H_ + H_)     // 1024 : [h0  | h1]

#define A0S  (K0 + 8)      // LDS row stride in halves (pad -> conflict-free)
#define A1S  (K1 + 8)
#define ZS   (H_ + 4)
#define CS   (H_ + 4)

typedef __attribute__((ext_vector_type(16))) _Float16 v16h;
typedef __attribute__((ext_vector_type(8)))  _Float16 v8h;
typedef __attribute__((ext_vector_type(8)))  float    v8f;

__device__ __forceinline__ float sigmoidf_(float x) { return 1.f / (1.f + __expf(-x)); }
__device__ __forceinline__ float tanhf_(float x)    { float t = __expf(2.f * x); return (t - 1.f) / (t + 1.f); }

// ---------------------------------------------------------------------------
// Core GEMM: D(16 x NT*16 per wave) += A(16xK, LDS f16) * B(KxN), B = packed
// weight rows (row n = output column n, K contiguous f16) streamed from L2.
// Fragment k order: contiguous, k = 16*hi + s, same in A and B (16-bit A 16x32; B 32x16).
// ---------------------------------------------------------------------------
template<int K, int NT>
__device__ __forceinline__ void wmma_gemm(const _Float16* __restrict__ As, int aStr,
                                          const _Float16* __restrict__ W, int nBase,
                                          v8f* acc)
{
    const int lane = threadIdx.x & 31;
    const int hi   = lane >> 4;        // K-half selector
    const int mr   = lane & 15;        // row (A) / column (B) within tile
    const _Float16* arow  = As + mr * aStr + hi * 16;
    const _Float16* brow0 = W + (size_t)(nBase + mr) * K + hi * 16;

    const v8f vzero = {0.f,0.f,0.f,0.f,0.f,0.f,0.f,0.f};
#pragma unroll
    for (int i = 0; i < NT; ++i) acc[i] = vzero;

#pragma unroll 2
    for (int kc = 0; kc < K; kc += 32) {
        v8h alo = *(const v8h*)(arow + kc);        // K: kc+hi*16  .. +8
        v8h ahi = *(const v8h*)(arow + kc + 8);    // K: kc+hi*16+8.. +8
        v16h a  = __builtin_shufflevector(alo, ahi, 0,1,2,3,4,5,6,7,8,9,10,11,12,13,14,15);
#pragma unroll
        for (int i = 0; i < NT; ++i) {
            v16h b = *(const v16h*)(brow0 + (size_t)i * 16 * K + kc);  // 32B contiguous
            acc[i] = __builtin_amdgcn_wmma_f32_16x16x32_f16(
                         false, a, false, b, (short)0, acc[i], false, false);
        }
    }
}

// ---------------------------------------------------------------------------
// LSTM cell epilogue, fully in-register: weights were packed so that column
// 4j+q = gate q of hidden unit j => lanes (base..base+3) of each acc VGPR hold
// (i,f,g,o) for one (m,j). Writer lane (lane%4==0) updates c and stores h(f16).
// ---------------------------------------------------------------------------
__device__ __forceinline__ void cell_update(const v8f* acc, const float* __restrict__ bp,
                                            float* cbuf,
                                            _Float16* h0p, int h0s,
                                            _Float16* h1p, int h1s)
{
    const int lane = threadIdx.x & 31;
    const int wave = threadIdx.x >> 5;
    const int base = lane & ~3;
#pragma unroll
    for (int i = 0; i < 8; ++i) {
        int col = wave * 128 + i * 16 + (lane & 15);
        float b = bp[col];
#pragma unroll
        for (int r = 0; r < 8; ++r) {
            float g  = acc[i][r] + b;
            float gi = __shfl(g, base + 0, 32);
            float gf = __shfl(g, base + 1, 32);
            float gg = __shfl(g, base + 2, 32);
            float go = __shfl(g, base + 3, 32);
            if ((lane & 3) == 0) {
                int m = r + ((lane >> 4) << 3);
                int j = col >> 2;
                float c = cbuf[m * CS + j];
                c = sigmoidf_(gf) * c + sigmoidf_(gi) * tanhf_(gg);
                cbuf[m * CS + j] = c;
                float h = sigmoidf_(go) * tanhf_(c);
                h0p[m * h0s + j] = (_Float16)h;
                if (h1p) h1p[m * h1s + j] = (_Float16)h;
            }
        }
    }
}

// ---------------------------------------------------------------------------
// Weight prep: fp32 -> f16, concat [Wih|Whh] along K, permute rows to
// gate-interleaved order (packed row 4j+q = original row q*512+j).
// ---------------------------------------------------------------------------
__global__ void prep_kernel(const float* __restrict__ Wih0, const float* __restrict__ Whh0,
                            const float* __restrict__ bih0, const float* __restrict__ bhh0,
                            const float* __restrict__ Wih1, const float* __restrict__ Whh1,
                            const float* __restrict__ bih1, const float* __restrict__ bhh1,
                            const float* __restrict__ fc1w,
                            _Float16* __restrict__ WB0, _Float16* __restrict__ WB1,
                            _Float16* __restrict__ WF1,
                            float* __restrict__ bp0, float* __restrict__ bp1)
{
    int i = blockIdx.x * blockDim.x + threadIdx.x;
    if (i < G_ * K0) {
        int n = i / K0, k = i - n * K0;
        int r0 = (n & 3) * H_ + (n >> 2);
        float v = (k < E_) ? Wih0[r0 * E_ + k] : Whh0[r0 * H_ + (k - E_)];
        WB0[i] = (_Float16)v;
    }
    if (i < G_ * K1) {
        int n = i >> 10, k = i & 1023;
        int r0 = (n & 3) * H_ + (n >> 2);
        float v = (k < H_) ? Wih1[r0 * H_ + k] : Whh1[r0 * H_ + (k - H_)];
        WB1[i] = (_Float16)v;
    }
    if (i < H_ * H_) WF1[i] = (_Float16)fc1w[i];
    if (i < G_) {
        int r0 = (i & 3) * H_ + (i >> 2);
        bp0[i] = bih0[r0] + bhh0[r0];
        bp1[i] = bih1[r0] + bhh1[r0];
    }
}

// ---------------------------------------------------------------------------
// Persistent kernel: each block owns BM=16 batch rows for the whole sequence.
// ---------------------------------------------------------------------------
__global__ __launch_bounds__(NTHREADS, 1)
void seq2seq_kernel(const float* __restrict__ x,
                    const float* __restrict__ We,  const float* __restrict__ be,
                    const _Float16* __restrict__ WB0, const _Float16* __restrict__ WB1,
                    const _Float16* __restrict__ WF1,
                    const float* __restrict__ bp0, const float* __restrict__ bp1,
                    const float* __restrict__ fc1b,
                    const float* __restrict__ lng, const float* __restrict__ lnb,
                    const float* __restrict__ fc2w, const float* __restrict__ fc2b,
                    float* __restrict__ out)
{
    __shared__ _Float16 A0[BM * A0S];   // [emb(256) | h0(512)] f16
    __shared__ _Float16 A1[BM * A1S];   // [h0(512)  | h1(512)] f16
    __shared__ float    c0s[BM * CS];
    __shared__ float    c1s[BM * CS];
    __shared__ float    zb[BM * ZS];    // fc1/LN scratch
    __shared__ float    xts[BM * FIN_]; // decoder feedback input

    const int tid  = threadIdx.x;
    const int lane = tid & 31;
    const int wave = tid >> 5;
    const int n0   = blockIdx.x * BM;

    // zero-init state
    for (int i = tid; i < BM * A0S; i += NTHREADS) A0[i] = (_Float16)0.f;
    for (int i = tid; i < BM * A1S; i += NTHREADS) A1[i] = (_Float16)0.f;
    for (int i = tid; i < BM * CS;  i += NTHREADS) { c0s[i] = 0.f; c1s[i] = 0.f; }
    __syncthreads();

    auto lstm_step = [&]() {
        v8f acc[8];
        wmma_gemm<K0, 8>(A0, A0S, WB0, wave * 128, acc);
        __syncthreads();                                   // all A0 reads done
        cell_update(acc, bp0, c0s, A0 + E_, A0S, A1, A1S); // h0 -> A0 tail, A1 head
        __syncthreads();
        wmma_gemm<K1, 8>(A1, A1S, WB1, wave * 128, acc);
        __syncthreads();
        cell_update(acc, bp1, c1s, A1 + H_, A1S, (_Float16*)nullptr, 0);
        __syncthreads();
    };

    // ---------------- encoder: t = 0..62 ----------------
    for (int t = 0; t < T_ - 1; ++t) {
#pragma unroll
        for (int s = 0; s < (BM * E_) / NTHREADS; ++s) {   // embed (F_IN=8, VALU)
            int idx = tid + s * NTHREADS;
            int m = idx >> 8, e = idx & (E_ - 1);
            const float* xr = x + ((size_t)(n0 + m) * T_ + t) * FIN_;
            const float* wr = We + e * FIN_;
            float a = be[e];
#pragma unroll
            for (int k = 0; k < FIN_; ++k) a = fmaf(xr[k], wr[k], a);
            A0[m * A0S + e] = (_Float16)a;
        }
        __syncthreads();
        lstm_step();
    }

    // ---------------- decoder: t = 0..31 ----------------
    if (tid < BM * FIN_) {
        int m = tid >> 3, f = tid & 7;
        xts[tid] = x[((size_t)(n0 + m) * T_ + (T_ - 1)) * FIN_ + f];
    }
    __syncthreads();

    for (int t = 0; t < FW_; ++t) {
#pragma unroll
        for (int s = 0; s < (BM * E_) / NTHREADS; ++s) {   // embed from feedback
            int idx = tid + s * NTHREADS;
            int m = idx >> 8, e = idx & (E_ - 1);
            const float* xr = xts + m * FIN_;
            const float* wr = We + e * FIN_;
            float a = be[e];
#pragma unroll
            for (int k = 0; k < FIN_; ++k) a = fmaf(xr[k], wr[k], a);
            A0[m * A0S + e] = (_Float16)a;
        }
        __syncthreads();
        lstm_step();

        // fc1: z = h1 @ fc1_w^T + b   (WMMA, 2 tiles/wave)
        {
            v8f acc2[2];
            wmma_gemm<H_, 2>(A1 + H_, A1S, WF1, wave * 32, acc2);
#pragma unroll
            for (int i = 0; i < 2; ++i) {
                int col = wave * 32 + i * 16 + (lane & 15);
                float b = fc1b[col];
#pragma unroll
                for (int r = 0; r < 8; ++r)
                    zb[(r + ((lane >> 4) << 3)) * ZS + col] = acc2[i][r] + b;
            }
        }
        __syncthreads();

        // LayerNorm + ReLU: one wave per batch row, shfl_xor reduction
        {
            int m = wave;
            float s1 = 0.f, s2 = 0.f;
#pragma unroll
            for (int q = 0; q < 16; ++q) {
                float v = zb[m * ZS + lane + q * 32];
                s1 += v; s2 += v * v;
            }
#pragma unroll
            for (int off = 16; off >= 1; off >>= 1) {
                s1 += __shfl_xor(s1, off, 32);
                s2 += __shfl_xor(s2, off, 32);
            }
            float mu  = s1 * (1.f / H_);
            float inv = rsqrtf(s2 * (1.f / H_) - mu * mu + 1e-5f);
#pragma unroll
            for (int q = 0; q < 16; ++q) {
                int c = lane + q * 32;
                float v = zb[m * ZS + c];
                v = (v - mu) * inv * lng[c] + lnb[c];
                zb[m * ZS + c] = fmaxf(v, 0.f);
            }
        }
        __syncthreads();

        // fc2 (512 -> 8) + output + feedback
        {
            int m = wave, f = lane >> 2, part = lane & 3;
            const float* wr = fc2w + f * H_ + part * 128;
            const float* zr = zb + m * ZS + part * 128;
            float s = 0.f;
#pragma unroll 8
            for (int k = 0; k < 128; ++k) s = fmaf(zr[k], wr[k], s);
            s += __shfl_xor(s, 1, 32);
            s += __shfl_xor(s, 2, 32);
            if (part == 0) {
                float y = s + fc2b[f];
                out[((size_t)(n0 + m) * FW_ + t) * FIN_ + f] = y;
                xts[m * FIN_ + f] = y;
            }
        }
        __syncthreads();
    }
}

extern "C" void kernel_launch(void* const* d_in, const int* in_sizes, int n_in,
                              void* d_out, int out_size, void* d_ws, size_t ws_size,
                              hipStream_t stream)
{
    (void)in_sizes; (void)n_in; (void)out_size; (void)ws_size;
    const float* x    = (const float*)d_in[0];
    const float* We   = (const float*)d_in[1];
    const float* be   = (const float*)d_in[2];
    const float* Wih0 = (const float*)d_in[3];
    const float* Whh0 = (const float*)d_in[4];
    const float* bih0 = (const float*)d_in[5];
    const float* bhh0 = (const float*)d_in[6];
    const float* Wih1 = (const float*)d_in[7];
    const float* Whh1 = (const float*)d_in[8];
    const float* bih1 = (const float*)d_in[9];
    const float* bhh1 = (const float*)d_in[10];
    const float* fc1w = (const float*)d_in[11];
    const float* fc1b = (const float*)d_in[12];
    const float* lng  = (const float*)d_in[13];
    const float* lnb  = (const float*)d_in[14];
    const float* fc2w = (const float*)d_in[15];
    const float* fc2b = (const float*)d_in[16];

    char* ws = (char*)d_ws;
    _Float16* WB0 = (_Float16*)(ws);                       // 2048*768  f16 = 3,145,728 B
    _Float16* WB1 = (_Float16*)(ws + 3145728);             // 2048*1024 f16 = 4,194,304 B
    _Float16* WF1 = (_Float16*)(ws + 7340032);             // 512*512   f16 =   524,288 B
    float*    bp0 = (float*)   (ws + 7864320);             // 2048 f32
    float*    bp1 = (float*)   (ws + 7872512);             // 2048 f32

    const int prepTotal = G_ * K1;                         // largest region
    prep_kernel<<<(prepTotal + 255) / 256, 256, 0, stream>>>(
        Wih0, Whh0, bih0, bhh0, Wih1, Whh1, bih1, bhh1, fc1w,
        WB0, WB1, WF1, bp0, bp1);

    seq2seq_kernel<<<N_ / BM, NTHREADS, 0, stream>>>(
        x, We, be, WB0, WB1, WF1, bp0, bp1, fc1b, lng, lnb, fc2w, fc2b,
        (float*)d_out);
}
